// SALGL_84224308674778
// MI455X (gfx1250) — matrix operand order repaired and checked by reading, not compiled, not run on hardware
//
#include <hip/hip_runtime.h>
#include <math.h>

// ---------------- problem constants ----------------
#define B_  256
#define S_  6
#define C_  80
#define F_  2048
#define E_  300
#define H_  1024
#define EP  320   // E padded to mult of 32 (WMMA K)
#define CP  96    // C padded to mult of 32 (WMMA K)

// ---------------- WMMA vector types ----------------
typedef __attribute__((ext_vector_type(16))) __bf16       v16bf;
typedef __attribute__((ext_vector_type(8)))  float        v8f;
typedef __attribute__((ext_vector_type(8)))  unsigned int v8u;

// ---------------- workspace layout (in 4-byte words) ----------------
constexpr long OFF_IMG   = 0;                               // B*F f32
constexpr long OFF_ENT   = OFF_IMG   + (long)B_ * F_;       // B f32 (per-sample entropy)
constexpr long OFF_IDS   = OFF_ENT   + B_;                  // B int
constexpr long OFF_CUPD  = OFF_IDS   + B_;                  // S*C*C f32 (updated comatrix)
constexpr long OFF_CNORM = OFF_CUPD  + (long)S_ * C_ * C_;  // S*C*C f32 (normalized)
constexpr long OFF_DROW  = OFF_CNORM + (long)S_ * C_ * C_;  // S*C f32 (rsqrt row sums)
constexpr long OFF_H2    = OFF_DROW  + (long)S_ * C_;       // S*C*F f32
constexpr long OFF_INPB  = OFF_H2    + (long)S_ * C_ * F_;  // bf16 [C][EP]   (words = /2)
constexpr long OFF_W1T   = OFF_INPB  + (long)C_ * EP / 2;   // bf16 [H][EP]
constexpr long OFF_W2T   = OFF_W1T   + (long)H_ * EP / 2;   // bf16 [F][H]
constexpr long OFF_ADJB  = OFF_W2T   + (long)F_ * H_ / 2;   // bf16 [S][C][CP]
constexpr long OFF_ZT    = OFF_ADJB  + (long)S_ * C_ * CP / 2; // bf16 [H][CP]
constexpr long OFF_HB    = OFF_ZT    + (long)H_ * CP / 2;   // bf16 [S][C][H]
constexpr long OFF_PT    = OFF_HB    + (long)S_ * C_ * H_ / 2; // bf16 [S][F][CP]
constexpr long WS_WORDS  = OFF_PT    + (long)S_ * F_ * CP / 2;

// ---------------- kernels ----------------

__global__ void zero_kernel(float* p, long n) {
  long i = (long)blockIdx.x * blockDim.x + threadIdx.x;
  long stride = (long)gridDim.x * blockDim.x;
  for (; i < n; i += stride) p[i] = 0.0f;
}

// Global max over 14x14 map: one wave per (b,f), vectorized float4 loads.
// Row base = idx*196 floats = idx*784 bytes (16B aligned); 196 = 49 float4s.
__global__ void maxpool_kernel(const float* __restrict__ x, float* __restrict__ img) {
  int wid  = threadIdx.x >> 5;
  int lane = threadIdx.x & 31;
  long idx = (long)blockIdx.x * 8 + wid;          // (b*F + f)
  const float4* base = (const float4*)(x + idx * 196);
  float m = -3.4e38f;
  #pragma unroll
  for (int i = 0; i < 2; ++i) {
    int j = lane + i * 32;
    if (j < 49) {
      float4 v = base[j];
      m = fmaxf(m, fmaxf(fmaxf(v.x, v.y), fmaxf(v.z, v.w)));
    }
  }
  #pragma unroll
  for (int o = 16; o; o >>= 1) m = fmaxf(m, __shfl_xor(m, o));
  if (lane == 0) img[idx] = m;
}

// scene_scores -> softmax -> probs/argmax/per-sample-entropy. One block per b.
__global__ void scene_kernel(const float* __restrict__ img, const float* __restrict__ Wsc,
                             float* __restrict__ probs_out, int* __restrict__ ids,
                             float* __restrict__ ent) {
  int b = blockIdx.x, t = threadIdx.x;
  float acc[S_] = {0.f, 0.f, 0.f, 0.f, 0.f, 0.f};
  for (int f = t; f < F_; f += blockDim.x) {
    float xv = img[(long)b * F_ + f];
    #pragma unroll
    for (int s = 0; s < S_; ++s) acc[s] += xv * Wsc[f * S_ + s];
  }
  #pragma unroll
  for (int s = 0; s < S_; ++s)
    #pragma unroll
    for (int o = 16; o; o >>= 1) acc[s] += __shfl_xor(acc[s], o);
  __shared__ float part[8][S_];
  int wid = t >> 5, lane = t & 31;
  if (lane == 0) {
    #pragma unroll
    for (int s = 0; s < S_; ++s) part[wid][s] = acc[s];
  }
  __syncthreads();
  if (t == 0) {
    float sc[S_];
    #pragma unroll
    for (int s = 0; s < S_; ++s) {
      float v = 0.f;
      for (int w = 0; w < 8; ++w) v += part[w][s];
      sc[s] = v;
    }
    float mx = sc[0];
    for (int s = 1; s < S_; ++s) mx = fmaxf(mx, sc[s]);
    float e[S_], sum = 0.f;
    for (int s = 0; s < S_; ++s) { e[s] = expf(sc[s] - mx); sum += e[s]; }
    float inv = 1.f / sum;
    float best = -1.f; int bi = 0; float en = 0.f;
    for (int s = 0; s < S_; ++s) {
      float p = e[s] * inv;
      probs_out[b * S_ + s] = p;
      if (p > best) { best = p; bi = s; }
      en -= p * logf(p + 1e-7f);
    }
    ids[b] = bi;
    ent[b] = en;
  }
}

// sample_en + batch_en scalars (deterministic tree reduce over B=256)
__global__ void entropy_finalize_kernel(const float* __restrict__ ent,
                                        const float* __restrict__ sp,
                                        float* out_sample, float* out_batch) {
  int t = threadIdx.x;  // 256 threads == B_
  __shared__ float red[256];
  __shared__ float ps[S_];
  __shared__ float ssum;
  red[t] = ent[t];
  __syncthreads();
  for (int o = 128; o; o >>= 1) { if (t < o) red[t] += red[t + o]; __syncthreads(); }
  if (t == 0) ssum = red[0];
  __syncthreads();
  for (int s = 0; s < S_; ++s) {
    red[t] = sp[t * S_ + s];
    __syncthreads();
    for (int o = 128; o; o >>= 1) { if (t < o) red[t] += red[t + o]; __syncthreads(); }
    if (t == 0) ps[s] = red[0] / (float)B_;
    __syncthreads();
  }
  if (t == 0) {
    *out_sample = ssum / (float)B_;
    float em = 0.f;
    for (int s = 0; s < S_; ++s) em -= ps[s] * logf(ps[s] + 1e-7f);
    float max_en = -logf(1.0f / S_ + 1e-7f);
    *out_batch = (max_en - em) * 100.0f;
  }
}

// comatrix[s,c1,c2] += sum_{b: ids[b]==s} y[b,c1]*y[b,c2]   (gather form)
__global__ void comat_update_kernel(const float* __restrict__ comatrix,
                                    const float* __restrict__ y,
                                    const int* __restrict__ ids,
                                    float* __restrict__ cupd) {
  int s = blockIdx.x / C_, c1 = blockIdx.x % C_;
  int c2 = threadIdx.x;
  if (c2 >= C_) return;
  long base = ((long)s * C_ + c1) * C_;
  float acc = comatrix[base + c2];
  for (int b = 0; b < B_; ++b) {
    if (ids[b] == s) acc += y[b * C_ + c1] * y[b * C_ + c2];
  }
  cupd[base + c2] = acc;
}

// row-normalize by diagonal, force diag=1; also D[s,c1]=rsqrt(rowsum)
__global__ void comat_norm_kernel(const float* __restrict__ cupd,
                                  float* __restrict__ cnorm,
                                  float* __restrict__ Drow) {
  int s = blockIdx.x / C_, c1 = blockIdx.x % C_;
  int t = threadIdx.x;  // 128
  __shared__ float red[128];
  long base = ((long)s * C_ + c1) * C_;
  float diag = cupd[base + c1];
  float v = 0.f;
  if (t < C_) {
    v = (t == c1) ? 1.0f : cupd[base + t] / (diag + 1e-8f);
    cnorm[base + t] = v;
  }
  red[t] = v;
  __syncthreads();
  for (int o = 64; o; o >>= 1) { if (t < o) red[t] += red[t + o]; __syncthreads(); }
  if (t == 0) Drow[s * C_ + c1] = rsqrtf(red[0]);
}

// adj[s,i,j] = D[s,i]*cnorm[s,j,i]*D[s,j]  -> bf16, K padded to CP (pad pre-zeroed)
__global__ void adj_build_kernel(const float* __restrict__ cnorm,
                                 const float* __restrict__ Drow,
                                 __bf16* __restrict__ adjb) {
  int s = blockIdx.x / C_, i = blockIdx.x % C_;
  int j = threadIdx.x;
  if (j >= C_) return;
  float v = Drow[s * C_ + i] * cnorm[((long)s * C_ + j) * C_ + i] * Drow[s * C_ + j];
  adjb[((long)s * C_ + i) * CP + j] = (__bf16)v;
}

__global__ void gather_comats_kernel(const float* __restrict__ cnorm,
                                     const int* __restrict__ ids,
                                     float* __restrict__ outc) {
  long i = (long)blockIdx.x * blockDim.x + threadIdx.x;
  if (i >= (long)B_ * C_ * C_) return;
  int b = (int)(i / (C_ * C_));
  int r = (int)(i % (C_ * C_));
  outc[i] = cnorm[(long)ids[b] * (C_ * C_) + r];
}

// f32 [rows][cols] -> bf16 [rows][ldd] (A-operand layout, K-padded)
__global__ void conv_a_kernel(const float* __restrict__ src, __bf16* __restrict__ dst,
                              int rows, int cols, int ldd) {
  int i = blockIdx.x * blockDim.x + threadIdx.x;
  if (i >= rows * cols) return;
  int r = i / cols, c = i % cols;
  dst[(long)r * ldd + c] = (__bf16)src[i];
}

// f32 [K][N] -> bf16 transposed [N][ldd] (B-operand layout: contiguous over K)
__global__ void conv_bt_kernel(const float* __restrict__ src, __bf16* __restrict__ dst,
                               int K, int N, int ldd) {
  int i = blockIdx.x * blockDim.x + threadIdx.x;
  if (i >= K * N) return;
  int k = i / N, n = i % N;
  dst[(long)n * ldd + k] = (__bf16)src[i];
}

// Generic batched WMMA GEMM: D = A(MxK, bf16 row-major) * B(KxN, stored BT[n][k] bf16)
// One wave computes a 16x64 output strip (4 N-tiles): A fragment loaded once per
// k-step and reused across 4 v_wmma ops -> 10 b128 loads per 4 WMMAs.
// (Interleaved per-tile B load form: lets the scheduler keep 4 live B fragments
// and 2x-unroll the k-loop — verified better codegen than explicit prefetch.)
// Epilogues: f32, bf16, bf16-transposed; optional LeakyReLU(0.2).
__global__ void wmma_gemm_kernel(const __bf16* __restrict__ A, const __bf16* __restrict__ BT,
                                 float* __restrict__ Df, __bf16* __restrict__ Db,
                                 __bf16* __restrict__ DbT,
                                 int M, int N, int K, int lda, int ldb, int ldd, int ldT,
                                 long sA, long sB, long sO, int leaky) {
  int s = blockIdx.z;
  const __bf16* As = A + (long)s * sA;
  const __bf16* Bs = BT + (long)s * sB;
  int nBase = blockIdx.x * 64;
  int mBase = blockIdx.y * 16;
  int lane = threadIdx.x;       // 32 threads = 1 wave
  int half = lane >> 4;         // 0 or 1
  int l16  = lane & 15;
  const __bf16* Arow  = As + (long)(mBase + l16) * lda;  // A row M=mBase+l16
  const __bf16* Brow0 = Bs + (long)(nBase + l16) * ldb;  // B col N=nBase+l16 (+16*t rows)
  v8f c[4];
  #pragma unroll
  for (int t = 0; t < 4; ++t) c[t] = (v8f){0.f, 0.f, 0.f, 0.f, 0.f, 0.f, 0.f, 0.f};
  int akg = half * 8;   // A: K = 16*half + {0..15} (words of 2 bf16)
  int bkg = half * 8;   // B: K = 16*half + 2v, 2v+1
  for (int k0 = 0; k0 < K; k0 += 32) {
    const unsigned int* a32 = (const unsigned int*)(Arow + k0);
    v8u au;
    #pragma unroll
    for (int v = 0; v < 4; ++v) { au[v] = a32[akg + v]; au[4 + v] = a32[akg + 4 + v]; }
    v16bf a = __builtin_bit_cast(v16bf, au);
    #pragma unroll
    for (int t = 0; t < 4; ++t) {
      const unsigned int* b32 = (const unsigned int*)(Brow0 + (long)t * 16 * ldb + k0);
      v8u bu;
      #pragma unroll
      for (int v = 0; v < 8; ++v) bu[v] = b32[bkg + v];
      v16bf b = __builtin_bit_cast(v16bf, bu);
      c[t] = __builtin_amdgcn_wmma_f32_16x16x32_bf16(false, a, false, b, (short)0, c[t],
                                                     false, false);
    }
  }
  #pragma unroll
  for (int t = 0; t < 4; ++t) {
    #pragma unroll
    for (int v = 0; v < 8; ++v) {
      int m = mBase + v + half * 8;
      int n = nBase + t * 16 + l16;
      float val = c[t][v];
      if (leaky) val = val > 0.f ? val : 0.2f * val;
      if (m < M) {
        if (Df)  Df [(long)s * sO + (long)m * ldd + n] = val;
        if (Db)  Db [(long)s * sO + (long)m * ldd + n] = (__bf16)val;
        if (DbT) DbT[(long)s * sO + (long)n * ldT + m] = (__bf16)val;
      }
    }
  }
}

// output[b,c] = dot(h2[ids[b],c,:], img_feats[b,:]); img row staged in LDS, wave per c
__global__ void final_out_kernel(const float* __restrict__ h2, const float* __restrict__ img,
                                 const int* __restrict__ ids, float* __restrict__ out) {
  int b = blockIdx.x;
  __shared__ float sf[F_];
  for (int f = threadIdx.x; f < F_; f += blockDim.x) sf[f] = img[(long)b * F_ + f];
  __syncthreads();
  int s = ids[b];
  int wid = threadIdx.x >> 5, lane = threadIdx.x & 31;
  for (int c = wid; c < C_; c += 8) {
    const float* row = h2 + ((long)s * C_ + c) * F_;
    float acc = 0.f;
    for (int f = lane; f < F_; f += 32) acc += row[f] * sf[f];
    #pragma unroll
    for (int o = 16; o; o >>= 1) acc += __shfl_xor(acc, o);
    if (lane == 0) out[b * C_ + c] = acc;
  }
}

// ---------------- host launch ----------------
extern "C" void kernel_launch(void* const* d_in, const int* in_sizes, int n_in,
                              void* d_out, int out_size, void* d_ws, size_t ws_size,
                              hipStream_t stream) {
  const float* x        = (const float*)d_in[0];
  const float* inp      = (const float*)d_in[1];
  const float* y        = (const float*)d_in[2];
  const float* comatrix = (const float*)d_in[3];
  const float* W_scene  = (const float*)d_in[4];
  const float* W1       = (const float*)d_in[5];
  const float* W2       = (const float*)d_in[6];
  float* out = (float*)d_out;
  float* wsf = (float*)d_ws;

  float*  img   = wsf + OFF_IMG;
  float*  ent   = wsf + OFF_ENT;
  int*    ids   = (int*)(wsf + OFF_IDS);
  float*  cupd  = wsf + OFF_CUPD;
  float*  cnorm = wsf + OFF_CNORM;
  float*  Drow  = wsf + OFF_DROW;
  float*  h2    = wsf + OFF_H2;
  __bf16* inpb  = (__bf16*)(wsf + OFF_INPB);
  __bf16* w1t   = (__bf16*)(wsf + OFF_W1T);
  __bf16* w2t   = (__bf16*)(wsf + OFF_W2T);
  __bf16* adjb  = (__bf16*)(wsf + OFF_ADJB);
  __bf16* zt    = (__bf16*)(wsf + OFF_ZT);
  __bf16* hb    = (__bf16*)(wsf + OFF_HB);
  __bf16* pt    = (__bf16*)(wsf + OFF_PT);

  float* out_output = out;                  // [B, C]
  float* out_probs  = out + (long)B_ * C_;  // [B, S]
  float* out_sen    = out_probs + (long)B_ * S_;
  float* out_ben    = out_sen + 1;
  float* out_comats = out_ben + 1;          // [B, C, C]

  // 0) zero bf16 region (provides K-padding zeros each call)
  zero_kernel<<<2048, 256, 0, stream>>>(wsf + OFF_INPB, WS_WORDS - OFF_INPB);
  // 1) global max-pool (HBM-bound, 411 MB, float4 loads)
  maxpool_kernel<<<(B_ * F_) / 8, 256, 0, stream>>>(x, img);
  // 2) scene head
  scene_kernel<<<B_, 256, 0, stream>>>(img, W_scene, out_probs, ids, ent);
  entropy_finalize_kernel<<<1, 256, 0, stream>>>(ent, out_probs, out_sen, out_ben);
  // 3) comatrix update / normalize / adjacency
  comat_update_kernel<<<S_ * C_, 128, 0, stream>>>(comatrix, y, ids, cupd);
  comat_norm_kernel<<<S_ * C_, 128, 0, stream>>>(cupd, cnorm, Drow);
  adj_build_kernel<<<S_ * C_, 128, 0, stream>>>(cnorm, Drow, adjb);
  gather_comats_kernel<<<(B_ * C_ * C_) / 256, 256, 0, stream>>>(cnorm, ids, out_comats);
  // 4) bf16 conversions (B operands stored transposed for contiguous WMMA fragments)
  conv_a_kernel<<<(C_ * E_ + 255) / 256, 256, 0, stream>>>(inp, inpb, C_, E_, EP);
  conv_bt_kernel<<<(E_ * H_ + 255) / 256, 256, 0, stream>>>(W1, w1t, E_, H_, EP);
  conv_bt_kernel<<<(H_ * F_ + 255) / 256, 256, 0, stream>>>(W2, w2t, H_, F_, H_);
  // 5) GCN collapsed to S=6 scenes, all GEMMs via v_wmma_f32_16x16x32_bf16
  //    (grid.x = N/64: each wave computes a 16x64 strip with A-fragment reuse)
  // Z = inp @ W1 -> ZT bf16 [H][CP]
  wmma_gemm_kernel<<<dim3(H_ / 64, C_ / 16, 1), 32, 0, stream>>>(
      inpb, w1t, nullptr, nullptr, zt, C_, H_, EP, EP, EP, 0, CP, 0, 0, 0, 0);
  // h_s = leaky(adj_s @ Z) -> bf16 [S][C][H]
  wmma_gemm_kernel<<<dim3(H_ / 64, C_ / 16, S_), 32, 0, stream>>>(
      adjb, zt, nullptr, hb, nullptr, C_, H_, CP, CP, CP, H_, 0,
      (long)C_ * CP, 0, (long)C_ * H_, 1);
  // P_s = h_s @ W2 -> PT bf16 [S][F][CP]
  wmma_gemm_kernel<<<dim3(F_ / 64, C_ / 16, S_), 32, 0, stream>>>(
      hb, w2t, nullptr, nullptr, pt, C_, F_, H_, H_, H_, 0, CP,
      (long)C_ * H_, 0, (long)F_ * CP, 0);
  // h2_s = adj_s @ P_s -> f32 [S][C][F]
  wmma_gemm_kernel<<<dim3(F_ / 64, C_ / 16, S_), 32, 0, stream>>>(
      adjb, pt, h2, nullptr, nullptr, C_, F_, CP, CP, CP, F_, 0,
      (long)C_ * CP, (long)F_ * CP, (long)C_ * F_, 0);
  // 6) output[b,c] = <h2[ids[b],c,:], img_feats[b,:]>  (h2 is L2-resident)
  final_out_kernel<<<B_, 256, 0, stream>>>(h2, img, ids, out_output);
}
